// Encoder_78795470012907
// MI455X (gfx1250) — matrix operand order repaired and checked by reading, not compiled, not run on hardware
//
#include <hip/hip_runtime.h>
#include <hip/hip_bf16.h>
#include <math.h>

// ---------------------------------------------------------------------------
// CDNA5 (gfx1250) wave32 WMMA types & helpers
// ---------------------------------------------------------------------------
typedef __attribute__((ext_vector_type(16))) __bf16 v16bf;
typedef __attribute__((ext_vector_type(8)))  __bf16 v8bf;
typedef __attribute__((ext_vector_type(8)))  float  v8f;

#define S_LEN 2048
#define BATCH 4
#define DMODEL 512
#define HEADS 8
#define DHEAD 64
#define MTOT (S_LEN * BATCH)   // 8192

__device__ __forceinline__ v8f wmma_bf16(v16bf a, v16bf b, v8f c) {
  // D = A(16x32 bf16) * B(32x16 bf16) + C(16x16 f32)
  return __builtin_amdgcn_wmma_f32_16x16x32_bf16(
      /*neg_a=*/false, a, /*neg_b=*/false, b,
      /*c_mod=*/(short)0, c, /*reuse_a=*/false, /*reuse_b=*/false);
}

// Combine two 8-element (16B) bf16 segments into one 16-element operand.
__device__ __forceinline__ v16bf make_ab(const __bf16* lo, const __bf16* hi) {
  v8bf a = *(const v8bf*)lo;
  v8bf b = *(const v8bf*)hi;
  v16bf r;
#pragma unroll
  for (int i = 0; i < 8; ++i) { r[i] = a[i]; r[8 + i] = b[i]; }
  return r;
}

// A-operand (16x32, 16-bit): lane holds row M=lane%16; lanes 0-15 carry
// K = {0..7, 16..23}; lanes 16-31 carry K = {8..15, 24..31}.
__device__ __forceinline__ v16bf load_a16(const __bf16* row, int half) {
  return make_ab(row + half * 8, row + 16 + half * 8);
}
// A-operand built from an f32 row (converted on the fly).
__device__ __forceinline__ v16bf load_a_f32(const float* row, int half) {
  const float* p0 = row + half * 8;
  const float* p1 = row + 16 + half * 8;
  v16bf r;
#pragma unroll
  for (int i = 0; i < 8; ++i) { r[i] = (__bf16)p0[i]; r[8 + i] = (__bf16)p1[i]; }
  return r;
}
// B-operand (32x16, 16-bit): lane holds col N=lane%16; lanes 0-15 hold
// K = {0..7, 16..23}; lanes 16-31 hold K = {8..15, 24..31} (matches A).
__device__ __forceinline__ v16bf load_b16(const __bf16* row, int half) {
  return make_ab(row + half * 8, row + 16 + half * 8);
}

// ---------------------------------------------------------------------------
// Kernel 1: QKV projection.  [MTOT, D] x [D, DHEAD] per (head, {Q,K,V}).
//   X rows m = s*B + b  (X is [S,B,D] flat).
//   Q stored PRE-SCALED by 1/sqrt(DK) as [H][B][S][DHEAD] bf16.
//   K stored as [H][B][S][DHEAD] bf16.
//   V stored transposed as [H][B][DHEAD][S] bf16 (A-operand friendly).
// ---------------------------------------------------------------------------
__global__ __launch_bounds__(128) void qkv_proj_kernel(
    const float* __restrict__ X,
    const float* __restrict__ WQ, const float* __restrict__ bQ,
    const float* __restrict__ WK, const float* __restrict__ bK,
    const float* __restrict__ WVp, const float* __restrict__ bVp,
    __bf16* __restrict__ Qb, __bf16* __restrict__ Kb, __bf16* __restrict__ Vt)
{
  __shared__ __bf16 Wt[64 * 32];  // weight chunk, transposed: [n(64)][kk(32)]

  const int lane = threadIdx.x & 31;
  const int wv   = threadIdx.x >> 5;
  const int half = lane >> 4;
  const int col  = lane & 15;
  const int h    = blockIdx.y;
  const int qkv  = blockIdx.z;
  const int m0   = blockIdx.x * 64 + wv * 16;

  const float* W    = (qkv == 0 ? WQ : qkv == 1 ? WK : WVp) + (size_t)h * DMODEL * DHEAD;
  const float* bias = (qkv == 0 ? bQ : qkv == 1 ? bK : bVp) + h * DHEAD;
  const float* xrow = X + (size_t)(m0 + col) * DMODEL;

  v8f zero = {0.f, 0.f, 0.f, 0.f, 0.f, 0.f, 0.f, 0.f};
  v8f acc[4];
#pragma unroll
  for (int i = 0; i < 4; ++i) acc[i] = zero;

  for (int kc = 0; kc < DMODEL / 32; ++kc) {
    __syncthreads();
    // Cooperative load of W[kc*32 .. +32][0..64] f32 -> LDS bf16, transposed.
    for (int i = threadIdx.x; i < 2048; i += 128) {
      int kk = i >> 6, n = i & 63;
      Wt[n * 32 + kk] = (__bf16)W[(size_t)(kc * 32 + kk) * DHEAD + n];
    }
    __syncthreads();

    v16bf a = load_a_f32(xrow + kc * 32, half);
#pragma unroll
    for (int sub = 0; sub < 4; ++sub) {
      v16bf b = load_b16(&Wt[(sub * 16 + col) * 32], half);
      acc[sub] = wmma_bf16(a, b, acc[sub]);
    }
  }

  // Bias add + scatter-store; hoist the uniform qkv branch OUTSIDE the loops
  // so the inner stores are branch-free.
  if (qkv == 2) {
#pragma unroll
    for (int sub = 0; sub < 4; ++sub) {
      float bb = bias[sub * 16 + col];
#pragma unroll
      for (int r = 0; r < 8; ++r) {
        int m = m0 + r + 8 * half;          // C layout: M = r + 8*(lane/16)
        int s = m >> 2, bi = m & 3;
        int n = sub * 16 + col;
        Vt[((size_t)(h * 4 + bi) * DHEAD + n) * S_LEN + s] =
            (__bf16)(acc[sub][r] + bb);
      }
    }
  } else {
    __bf16* dst   = (qkv == 0) ? Qb : Kb;
    const float scale = (qkv == 0) ? 0.125f : 1.0f;  // fold 1/sqrt(DK) into Q
#pragma unroll
    for (int sub = 0; sub < 4; ++sub) {
      float bb = bias[sub * 16 + col];
#pragma unroll
      for (int r = 0; r < 8; ++r) {
        int m = m0 + r + 8 * half;
        int s = m >> 2, bi = m & 3;
        int n = sub * 16 + col;
        dst[((size_t)(h * 4 + bi) * S_LEN + s) * DHEAD + n] =
            (__bf16)((acc[sub][r] + bb) * scale);
      }
    }
  }
}

// ---------------------------------------------------------------------------
// Kernel 2: fused attention with the double softmax.
//   Grid (S/16, H); block = 128 = 4 waves, wave w owns batch b = w.
//   Stream over t in chunks of 32:
//     scores = Qs(16x64) * K^T            (4 bf16 WMMAs; Q pre-scaled)
//     e = exp(score)  (scores ~ N(0,1): no max needed)  -> LDS exchange
//     p1 = e_b / sum_b e   (softmax over batch)
//     p  = exp(p1)  in [1,e]  -> pbf (same-wave LDS transpose buffer)
//     Z^T(64x16) += V^T(16x32) * P^T(32x16)             (4 bf16 WMMAs)
//   e-exchange is parity double-buffered -> ONE barrier per chunk.
//   Finish: divide by row sums, store Zc[m][h*64+v] bf16.
// ---------------------------------------------------------------------------
__global__ __launch_bounds__(128) void attn_kernel(
    const __bf16* __restrict__ Qb, const __bf16* __restrict__ Kb,
    const __bf16* __restrict__ Vt, __bf16* __restrict__ Zc)
{
  __shared__ float  ebuf[2][BATCH][16][32];  // exp(score), all batches (16KB)
  __shared__ __bf16 pbf[BATCH][16][32];      // per-wave P transpose staging
  __shared__ float  rsl[BATCH][16];          // per-row denominators

  const int lane = threadIdx.x & 31;
  const int b    = threadIdx.x >> 5;     // wave -> batch
  const int half = lane >> 4;
  const int col  = lane & 15;
  const int h    = blockIdx.y;
  const int s0   = blockIdx.x * 16;
  const size_t hb = (size_t)(h * 4 + b);

  const __bf16* Kbase = Kb + hb * S_LEN * DHEAD;
  const __bf16* Vbase = Vt + hb * DHEAD * S_LEN;

  // Q A-operands for this wave's 16 s-rows (dk chunks 0..31, 32..63).
  const __bf16* qrow = Qb + (hb * S_LEN + s0 + col) * DHEAD;
  v16bf aq0 = load_a16(qrow, half);
  v16bf aq1 = load_a16(qrow + 32, half);

  v8f zero = {0.f, 0.f, 0.f, 0.f, 0.f, 0.f, 0.f, 0.f};
  v8f zacc[4];
#pragma unroll
  for (int i = 0; i < 4; ++i) zacc[i] = zero;
  float rsum[8] = {0.f, 0.f, 0.f, 0.f, 0.f, 0.f, 0.f, 0.f};

  int par = 0;
  for (int t0 = 0; t0 < S_LEN; t0 += 32, par ^= 1) {
    // ---- prefetch next chunk's K/V streams (global_prefetch_b8) ----
    if (t0 + 32 < S_LEN) {
      __builtin_prefetch(Kbase + (size_t)(t0 + 32 + lane) * DHEAD, 0, 0);
      __builtin_prefetch(Vbase + (size_t)lane * S_LEN + t0 + 32, 0, 0);
      __builtin_prefetch(Vbase + (size_t)(32 + lane) * S_LEN + t0 + 32, 0, 0);
    }

    // ---- scores (two 16x16 t-subtiles) + own-batch exponentials ----
    float ereg[2][8];
#pragma unroll
    for (int sub = 0; sub < 2; ++sub) {
      const __bf16* krow = Kbase + (size_t)(t0 + sub * 16 + col) * DHEAD;
      v8f a = zero;
      a = wmma_bf16(aq0, load_b16(krow, half), a);        // dk 0..31
      a = wmma_bf16(aq1, load_b16(krow + 32, half), a);   // dk 32..63
#pragma unroll
      for (int r = 0; r < 8; ++r) {
        float e = __expf(a[r]);                 // scores ~N(0,1): safe
        ereg[sub][r] = e;
        ebuf[par][b][r + 8 * half][sub * 16 + col] = e;
      }
    }
    __syncthreads();   // the only cross-wave exchange point per chunk

    // ---- softmax over batch, second-level exp, stage P^T ----
#pragma unroll
    for (int sub = 0; sub < 2; ++sub) {
#pragma unroll
      for (int r = 0; r < 8; ++r) {
        int si = r + 8 * half, ti = sub * 16 + col;
        float den = ebuf[par][0][si][ti] + ebuf[par][1][si][ti] +
                    ebuf[par][2][si][ti] + ebuf[par][3][si][ti];
        float p = __expf(ereg[sub][r] / den);   // in [1,e]: no rescale needed
        rsum[r] += p;                           // second softmax denominator
        pbf[b][si][ti] = (__bf16)p;             // same-wave buffer (in-order LDS)
      }
    }

    // ---- Z^T += V^T * P^T  (pbf produced & consumed by this wave only) ----
    v16bf bp = load_b16(&pbf[b][col][0], half);  // B: [t(32) x s(16)]
#pragma unroll
    for (int vs = 0; vs < 4; ++vs) {
      const __bf16* vrow = Vbase + (size_t)(vs * 16 + col) * S_LEN + t0;
      v16bf av = load_a16(vrow, half);           // A: [v(16) x t(32)]
      zacc[vs] = wmma_bf16(av, bp, zacc[vs]);
    }
  }

  // ---- reduce row sums across the 16 t-columns of each lane group ----
#pragma unroll
  for (int r = 0; r < 8; ++r) {
    float v = rsum[r];
    v += __shfl_xor(v, 1, 32);
    v += __shfl_xor(v, 2, 32);
    v += __shfl_xor(v, 4, 32);
    v += __shfl_xor(v, 8, 32);
    if (col == 0) rsl[b][r + 8 * half] = v;
  }
  __syncthreads();
  float inv = 1.0f / rsl[b][col];   // zacc C layout: N = s-col = lane%16

  // ---- store Zc[m][h*64 + v] : 8 consecutive v per lane -> 16B store ----
  const size_t m = (size_t)(s0 + col) * BATCH + b;
#pragma unroll
  for (int vs = 0; vs < 4; ++vs) {
    v8bf o;
#pragma unroll
    for (int r = 0; r < 8; ++r) o[r] = (__bf16)(zacc[vs][r] * inv);
    *(v8bf*)(Zc + m * (HEADS * DHEAD) + h * DHEAD + vs * 16 + 8 * half) = o;
  }
}

// ---------------------------------------------------------------------------
// Kernel 3: output projection.  out[m, d] = Zc[m, :] * WO[:, d] + bO[d].
//   m = s*B + b, so rows map directly onto d_out's [S,B,D] layout (f32).
// ---------------------------------------------------------------------------
__global__ __launch_bounds__(128) void out_proj_kernel(
    const __bf16* __restrict__ Zc, const float* __restrict__ WO,
    const float* __restrict__ bO, float* __restrict__ out)
{
  __shared__ __bf16 Wt[64 * 32];

  const int lane = threadIdx.x & 31;
  const int wv   = threadIdx.x >> 5;
  const int half = lane >> 4;
  const int col  = lane & 15;
  const int m0   = blockIdx.x * 64 + wv * 16;
  const int n0   = blockIdx.y * 64;

  const __bf16* zrow = Zc + (size_t)(m0 + col) * (HEADS * DHEAD);

  v8f zero = {0.f, 0.f, 0.f, 0.f, 0.f, 0.f, 0.f, 0.f};
  v8f acc[4];
#pragma unroll
  for (int i = 0; i < 4; ++i) acc[i] = zero;

  for (int kc = 0; kc < (HEADS * DHEAD) / 32; ++kc) {
    __syncthreads();
    for (int i = threadIdx.x; i < 2048; i += 128) {
      int kk = i >> 6, n = i & 63;
      Wt[n * 32 + kk] = (__bf16)WO[(size_t)(kc * 32 + kk) * DMODEL + n0 + n];
    }
    __syncthreads();

    v16bf a = load_a16(zrow + kc * 32, half);
#pragma unroll
    for (int sub = 0; sub < 4; ++sub) {
      v16bf b = load_b16(&Wt[(sub * 16 + col) * 32], half);
      acc[sub] = wmma_bf16(a, b, acc[sub]);
    }
  }

#pragma unroll
  for (int sub = 0; sub < 4; ++sub) {
    float bb = bO[n0 + sub * 16 + col];
#pragma unroll
    for (int r = 0; r < 8; ++r) {
      size_t m = (size_t)m0 + r + 8 * half;
      out[m * DMODEL + n0 + sub * 16 + col] = acc[sub][r] + bb;
    }
  }
}

// ---------------------------------------------------------------------------
extern "C" void kernel_launch(void* const* d_in, const int* in_sizes, int n_in,
                              void* d_out, int out_size, void* d_ws, size_t ws_size,
                              hipStream_t stream) {
  const float* X   = (const float*)d_in[0];
  const float* WQ  = (const float*)d_in[1];
  const float* bQ  = (const float*)d_in[2];
  const float* WK  = (const float*)d_in[3];
  const float* bK  = (const float*)d_in[4];
  const float* WVp = (const float*)d_in[5];
  const float* bVp = (const float*)d_in[6];
  const float* WO  = (const float*)d_in[7];
  const float* bO  = (const float*)d_in[8];
  float* out = (float*)d_out;

  char* ws = (char*)d_ws;
  const size_t qsz = (size_t)HEADS * BATCH * S_LEN * DHEAD * sizeof(__bf16); // 8.39 MB
  __bf16* Qb = (__bf16*)(ws);
  __bf16* Kb = (__bf16*)(ws + qsz);
  __bf16* Vt = (__bf16*)(ws + 2 * qsz);
  __bf16* Zc = (__bf16*)(ws + 3 * qsz);   // [MTOT][H*DV] bf16

  qkv_proj_kernel<<<dim3(MTOT / 64, HEADS, 3), 128, 0, stream>>>(
      X, WQ, bQ, WK, bK, WVp, bVp, Qb, Kb, Vt);
  attn_kernel<<<dim3(S_LEN / 16, HEADS), 128, 0, stream>>>(Qb, Kb, Vt, Zc);
  out_proj_kernel<<<dim3(MTOT / 64, DMODEL / 64), 128, 0, stream>>>(Zc, WO, bO, out);
}
